// MultiheadAttention_53137335386290
// MI455X (gfx1250) — matrix operand order repaired and checked by reading, not compiled, not run on hardware
//
#include <hip/hip_runtime.h>

typedef __attribute__((ext_vector_type(16))) __bf16 v16bf;
typedef __attribute__((ext_vector_type(8)))  float  v8f;
typedef __attribute__((ext_vector_type(4)))  int    v4i;

#define S_LEN  2048
#define HEADS  16
#define DKV    64
#define DMODEL 1024

#define G_STRIDE  72   // bf16 per LDS row for 64-wide chunks (144B: 16B-aligned, bank-stride 36)
#define KV_STRIDE 72
#define P_STRIDE  34   // bf16 per LDS row for P transpose slab

// ---------------------------------------------------------------------------
// gfx1250 async global->LDS staging (builtins confirmed present; typed v4i ptrs).
// ---------------------------------------------------------------------------
#if __has_builtin(__builtin_amdgcn_global_load_async_to_lds_b128) && \
    __has_builtin(__builtin_amdgcn_s_wait_asynccnt)
#define USE_ASYNC_LDS 1
#else
#define USE_ASYNC_LDS 0
#endif

__device__ __forceinline__ void copy16_g2l(const void* g, void* l) {
#if USE_ASYNC_LDS
    typedef __attribute__((address_space(1))) v4i* as1_v4i;
    typedef __attribute__((address_space(3))) v4i* as3_v4i;
    v4i* gt = (v4i*)g;
    v4i* lt = (v4i*)l;
    __builtin_amdgcn_global_load_async_to_lds_b128((as1_v4i)gt, (as3_v4i)lt, 0, 0);
#else
    *(uint4*)l = *(const uint4*)g;
#endif
}
__device__ __forceinline__ void async_stage_wait() {
#if USE_ASYNC_LDS
    __builtin_amdgcn_s_wait_asynccnt(0);
#endif
}

// K-pair base offset for 16-bit A-matrix fragment (16x32; M on lanes 0-15,
// upper half-lanes hold K+16..K+31, contiguous k = 16*half + 2*v).
__device__ __forceinline__ int a_kbase(int v, int half) {
    return 2 * v + 16 * half;
}

__device__ __forceinline__ unsigned pack2bf(float x, float y) {
    union { __bf16 h[2]; unsigned u; } c;
    c.h[0] = (__bf16)x; c.h[1] = (__bf16)y;
    return c.u;
}

// ---------------------------------------------------------------------------
// Block-tiled GEMM: Y[M,N] = A[M,K] @ W[N,K]^T + bias (bf16 WMMA, f32 accum).
// Block = 128(m) x 128(n); 8 waves in a 2x4 grid, each computing 64x32
// (4x2 WMMA tiles -> 8 WMMAs per 32-wide K step, 16 per barrier).
// K chunks of 64 double-buffered through LDS.
// mode 0: bf16 store into split-head layout [B, H, S, 64]
// mode 1: f32 row-major [M, N]
// ---------------------------------------------------------------------------
template <typename AT>
__global__ void __launch_bounds__(256)
gemm_nt(const AT* __restrict__ A, const float* __restrict__ W,
        const float* __restrict__ bias, void* __restrict__ dst,
        int M, int N, int K, int mode)
{
    __shared__ __bf16 Abuf[2][128 * G_STRIDE];
    __shared__ __bf16 Wbuf[2][128 * G_STRIDE];

    const int tid  = threadIdx.x;
    const int lane = tid & 31, wid = tid >> 5;
    const int half = lane >> 4, l16 = lane & 15;
    const int wm   = wid >> 2,  wn  = wid & 3;     // 2 x 4 wave grid

    const int nblk = N >> 7;                       // N / 128
    const int m0 = (blockIdx.x / nblk) * 128;
    const int n0 = (blockIdx.x % nblk) * 128;

    auto stageA = [&](int buf, int k0) {
        if constexpr (sizeof(AT) == 4) {           // fp32 -> bf16 on the fly
            const float* Af = (const float*)A;
#pragma unroll
            for (int i = 0; i < 8; ++i) {
                const int f = tid + i * 256;       // float4 id 0..2047
                const int row = f >> 4, c4 = (f & 15) * 4;
                const float4 x = *(const float4*)(Af + (size_t)(m0 + row) * K + k0 + c4);
                *(uint2*)&Abuf[buf][row * G_STRIDE + c4] =
                    make_uint2(pack2bf(x.x, x.y), pack2bf(x.z, x.w));
            }
        } else {                                   // bf16 input (ctx): async raw copy
#pragma unroll
            for (int i = 0; i < 4; ++i) {
                const int f = tid + i * 256;       // 16B unit id 0..1023
                const int row = f >> 3, c8 = (f & 7) * 8;
                copy16_g2l((const __bf16*)A + (size_t)(m0 + row) * K + k0 + c8,
                           &Abuf[buf][row * G_STRIDE + c8]);
            }
        }
    };
    auto stageW = [&](int buf, int k0) {
#pragma unroll
        for (int i = 0; i < 8; ++i) {
            const int f = tid + i * 256;           // float4 id 0..2047
            const int row = f >> 4, c4 = (f & 15) * 4;
            const float4 x = *(const float4*)(W + (size_t)(n0 + row) * K + k0 + c4);
            *(uint2*)&Wbuf[buf][row * G_STRIDE + c4] =
                make_uint2(pack2bf(x.x, x.y), pack2bf(x.z, x.w));
        }
    };

    v8f acc[4][2] = {};
    const int NK = K >> 6;                         // 64-wide chunks

    stageA(0, 0);
    stageW(0, 0);
    for (int kc = 0; kc < NK; ++kc) {
        async_stage_wait();
        __syncthreads();
        const int cur = kc & 1;
        if (kc + 1 < NK) { stageA(1 - cur, (kc + 1) << 6); stageW(1 - cur, (kc + 1) << 6); }

#pragma unroll
        for (int ks = 0; ks < 2; ++ks) {           // two 32-wide K steps
            v16bf af[4], bfr[2];
#pragma unroll
            for (int im = 0; im < 4; ++im) {
                const int r = wm * 64 + im * 16 + l16;
#pragma unroll
                for (int v = 0; v < 8; ++v) {
                    const int kb = ks * 32 + a_kbase(v, half);
                    af[im][2 * v]     = Abuf[cur][r * G_STRIDE + kb];
                    af[im][2 * v + 1] = Abuf[cur][r * G_STRIDE + kb + 1];
                }
            }
#pragma unroll
            for (int in = 0; in < 2; ++in) {
                const int r = wn * 32 + in * 16 + l16;
#pragma unroll
                for (int v = 0; v < 8; ++v) {
                    const int kk = ks * 32 + 2 * v + 16 * half;
                    bfr[in][2 * v]     = Wbuf[cur][r * G_STRIDE + kk];
                    bfr[in][2 * v + 1] = Wbuf[cur][r * G_STRIDE + kk + 1];
                }
            }
#pragma unroll
            for (int im = 0; im < 4; ++im)
#pragma unroll
                for (int in = 0; in < 2; ++in)
                    acc[im][in] = __builtin_amdgcn_wmma_f32_16x16x32_bf16(
                        false, af[im], false, bfr[in], (short)0, acc[im][in], false, false);
        }
    }

#pragma unroll
    for (int im = 0; im < 4; ++im) {
#pragma unroll
        for (int in = 0; in < 2; ++in) {
            const int col = n0 + wn * 32 + in * 16 + l16;
            const float bb = bias[col];
            if (mode == 0) {
                __bf16* out = (__bf16*)dst;
                const int h = col >> 6, d = col & 63;
#pragma unroll
                for (int v = 0; v < 8; ++v) {
                    const int m = m0 + wm * 64 + im * 16 + v + 8 * half;
                    const int bidx = m >> 11;           // m / S_LEN
                    const int s    = m & (S_LEN - 1);
                    out[(((size_t)bidx * HEADS + h) * S_LEN + s) * DKV + d] =
                        (__bf16)(acc[im][in][v] + bb);
                }
            } else {
                float* out = (float*)dst;
#pragma unroll
                for (int v = 0; v < 8; ++v) {
                    const int m = m0 + wm * 64 + im * 16 + v + 8 * half;
                    out[(size_t)m * N + col] = acc[im][in][v] + bb;
                }
            }
        }
    }
}

// ---------------------------------------------------------------------------
// Flash attention: one block = one (b, h) and 128 query rows (8 waves x 16).
// 64-key K/V tiles staged cooperatively into LDS (double-buffered, async),
// two 32-key WMMA sub-chunks per stage, online softmax, O accumulated as
// 4x f32 16x16 WMMA tiles per wave.
// ---------------------------------------------------------------------------
__global__ void __launch_bounds__(256)
flash_attn(const __bf16* __restrict__ qh, const __bf16* __restrict__ kh,
           const __bf16* __restrict__ vh, const int* __restrict__ mask,
           __bf16* __restrict__ ctx)
{
    __shared__ __bf16 Kbuf[2][64 * KV_STRIDE];
    __shared__ __bf16 Vbuf[2][64 * KV_STRIDE];
    __shared__ __bf16 Pbuf[8][16 * P_STRIDE];

    const int tid  = threadIdx.x;
    const int lane = tid & 31, wid = tid >> 5;
    const int half = lane >> 4, l16 = lane & 15;

    const int blocks_per_bh = S_LEN / 128;            // 16
    const int bh = blockIdx.x / blocks_per_bh;        // b*H + h
    const int b  = bh / HEADS, h = bh % HEADS;
    const int q0 = (blockIdx.x % blocks_per_bh) * 128 + wid * 16;

    const size_t hb = (size_t)bh * S_LEN * DKV;
    const __bf16* Q  = qh + hb;
    const __bf16* Kg = kh + hb;
    const __bf16* Vg = vh + hb;
    const int* Mrow = mask + ((size_t)b * S_LEN + q0) * S_LEN;

    auto stageKV = [&](int buf, int k0) {
#pragma unroll
        for (int i = 0; i < 2; ++i) {
            const int f = tid + i * 256;              // 16B unit id 0..511
            const int key = f >> 3, c8 = (f & 7) * 8; // 64 keys x 64 d
            copy16_g2l(Kg + (size_t)(k0 + key) * DKV + c8,
                       &Kbuf[buf][key * KV_STRIDE + c8]);
            copy16_g2l(Vg + (size_t)(k0 + key) * DKV + c8,
                       &Vbuf[buf][key * KV_STRIDE + c8]);
        }
    };

    // Q fragments (d_k = 64 -> two 16x32 A-frags), kept in registers.
    v16bf qf[2];
#pragma unroll
    for (int dc = 0; dc < 2; ++dc)
#pragma unroll
        for (int v = 0; v < 8; ++v) {
            const int kb = dc * 32 + a_kbase(v, half);
            const __bf16* p = Q + (size_t)(q0 + l16) * DKV + kb;
            qf[dc][2 * v]     = p[0];
            qf[dc][2 * v + 1] = p[1];
        }

    v8f o0 = {}, o1 = {}, o2 = {}, o3 = {};
    float run_m[8], run_l[8];
#pragma unroll
    for (int v = 0; v < 8; ++v) { run_m[v] = -3.0e38f; run_l[v] = 0.0f; }

    stageKV(0, 0);
    for (int kb64 = 0; kb64 < S_LEN; kb64 += 64) {
        async_stage_wait();
        __syncthreads();
        const int cur = (kb64 >> 6) & 1;
        if (kb64 + 64 < S_LEN) stageKV(1 - cur, kb64 + 64);

#pragma unroll
        for (int sc = 0; sc < 2; ++sc) {              // two 32-key sub-chunks
            const int kloc = sc * 32;

            // ---- scores: two 16x16 tiles covering 32 keys ----
            v8f s[2];
#pragma unroll
            for (int khf = 0; khf < 2; ++khf) {
                v8f acc = {};
                const int krow = kloc + khf * 16 + l16;
                const int key  = kb64 + krow;
#pragma unroll
                for (int dc = 0; dc < 2; ++dc) {
                    v16bf bfr;
#pragma unroll
                    for (int v = 0; v < 8; ++v) {
                        const int kk = dc * 32 + 2 * v + 16 * half;
                        bfr[2 * v]     = Kbuf[cur][krow * KV_STRIDE + kk];
                        bfr[2 * v + 1] = Kbuf[cur][krow * KV_STRIDE + kk + 1];
                    }
                    acc = __builtin_amdgcn_wmma_f32_16x16x32_bf16(
                        false, qf[dc], false, bfr, (short)0, acc, false, false);
                }
#pragma unroll
                for (int v = 0; v < 8; ++v) {
                    const int m = v + 8 * half;
                    const float scv = acc[v] * 0.125f;    // 1/sqrt(64)
                    acc[v] = (Mrow[(size_t)m * S_LEN + key] == 0) ? -1.0e9f : scv;
                }
                s[khf] = acc;
            }

            // ---- online softmax per query row ----
#pragma unroll
            for (int v = 0; v < 8; ++v) {
                float vmax = fmaxf(s[0][v], s[1][v]);
#pragma unroll
                for (int off = 8; off >= 1; off >>= 1)
                    vmax = fmaxf(vmax, __shfl_xor(vmax, off, 32));
                const float nm   = fmaxf(run_m[v], vmax);
                const float corr = __expf(run_m[v] - nm);
                const float p0 = __expf(s[0][v] - nm);
                const float p1 = __expf(s[1][v] - nm);
                float rs = p0 + p1;
#pragma unroll
                for (int off = 8; off >= 1; off >>= 1)
                    rs += __shfl_xor(rs, off, 32);
                run_l[v] = run_l[v] * corr + rs;
                run_m[v] = nm;
                o0[v] *= corr; o1[v] *= corr; o2[v] *= corr; o3[v] *= corr;
                const int m = v + 8 * half;
                Pbuf[wid][m * P_STRIDE + l16]      = (__bf16)p0;
                Pbuf[wid][m * P_STRIDE + 16 + l16] = (__bf16)p1;
            }
            __builtin_amdgcn_wave_barrier();   // same-wave LDS write -> read

            // ---- P (16x32) as A-frag from the per-wave LDS slab ----
            v16bf pa;
#pragma unroll
            for (int v = 0; v < 8; ++v) {
                const int kb = a_kbase(v, half);
                pa[2 * v]     = Pbuf[wid][l16 * P_STRIDE + kb];
                pa[2 * v + 1] = Pbuf[wid][l16 * P_STRIDE + kb + 1];
            }

            // ---- O += P @ V over 4 d-chunks of 16 ----
#pragma unroll
            for (int f = 0; f < 4; ++f) {
                v16bf bv_;
#pragma unroll
                for (int v = 0; v < 8; ++v) {
                    const int kk = kloc + 2 * v + 16 * half;
                    const int n  = f * 16 + l16;
                    bv_[2 * v]     = Vbuf[cur][kk * KV_STRIDE + n];
                    bv_[2 * v + 1] = Vbuf[cur][(kk + 1) * KV_STRIDE + n];
                }
                v8f* op = (f == 0) ? &o0 : (f == 1) ? &o1 : (f == 2) ? &o2 : &o3;
                *op = __builtin_amdgcn_wmma_f32_16x16x32_bf16(
                    false, pa, false, bv_, (short)0, *op, false, false);
            }
        }
    }

    // ---- normalize, write ctx [B, S, H*64] bf16 ----
#pragma unroll
    for (int f = 0; f < 4; ++f) {
        const v8f o = (f == 0) ? o0 : (f == 1) ? o1 : (f == 2) ? o2 : o3;
#pragma unroll
        for (int v = 0; v < 8; ++v) {
            const int m = q0 + v + 8 * half;
            const float val = o[v] / run_l[v];
            ctx[((size_t)b * S_LEN + m) * DMODEL + h * DKV + f * 16 + l16] =
                (__bf16)val;
        }
    }
}

// ---------------------------------------------------------------------------
extern "C" void kernel_launch(void* const* d_in, const int* in_sizes, int n_in,
                              void* d_out, int out_size, void* d_ws, size_t ws_size,
                              hipStream_t stream)
{
    const float* q    = (const float*)d_in[0];
    const float* k    = (const float*)d_in[1];
    const float* v    = (const float*)d_in[2];
    const int*   mask = (const int*)  d_in[3];
    const float* wq   = (const float*)d_in[4];
    const float* bq   = (const float*)d_in[5];
    const float* wk   = (const float*)d_in[6];
    const float* bk   = (const float*)d_in[7];
    const float* wv   = (const float*)d_in[8];
    const float* bv   = (const float*)d_in[9];
    const float* wo   = (const float*)d_in[10];
    const float* bo   = (const float*)d_in[11];

    const int M = 4 * S_LEN;            // 8192
    const int N = DMODEL, K = DMODEL;   // 1024

    const size_t elems = (size_t)M * DMODEL;   // 8M bf16 per tensor
    __bf16* qh  = (__bf16*)d_ws;
    __bf16* kh  = qh + elems;
    __bf16* vh  = kh + elems;
    __bf16* ctx = vh + elems;

    const int gemm_blocks = (M / 128) * (N / 128);     // 512

    gemm_nt<float><<<gemm_blocks, 256, 0, stream>>>(q, wq, bq, qh, M, N, K, 0);
    gemm_nt<float><<<gemm_blocks, 256, 0, stream>>>(k, wk, bk, kh, M, N, K, 0);
    gemm_nt<float><<<gemm_blocks, 256, 0, stream>>>(v, wv, bv, vh, M, N, K, 0);

    const int attn_blocks = 4 * HEADS * (S_LEN / 128); // 1024
    flash_attn<<<attn_blocks, 256, 0, stream>>>(qh, kh, vh, mask, ctx);

    gemm_nt<__bf16><<<gemm_blocks, 256, 0, stream>>>(ctx, wo, bo, d_out, M, N, K, 1);
}
